// GCN_w_86354612453998
// MI455X (gfx1250) — compile-verified, matrix operand order checked by reading, not run on hardware
//
#include <hip/hip_runtime.h>
#include <hip/hip_bf16.h>

typedef __attribute__((ext_vector_type(16))) __bf16   v16bf;
typedef __attribute__((ext_vector_type(8)))  float    v8f;
typedef __attribute__((ext_vector_type(8)))  unsigned v8u;

// Packed bf16 ReLU on 2 elements: v_pk_max_num_bf16 (CDNA5 VOP3P op 44).
// No clang builtin exposes this; inline asm.
__device__ __forceinline__ unsigned pk_relu_bf16(unsigned x) {
    unsigned r;
    asm("v_pk_max_num_bf16 %0, %1, 0" : "=v"(r) : "v"(x));
    return r;
}

// ---------------------------------------------------------------------------
// Pack fp32 weights (fi x fo, row-major) into WMMA B-fragment layout, bf16.
// Fragment index: ((kb * (fo/16) + tN) * 32 + lane) * 16 + e
//   element e of lane L holds W[kb*32 + kmap(L,e)][tN*16 + (L&15)]
//   kmap(L,e) = e + 8*(L>>4) + (e>=8 ? 8 : 0)   (ISA 7.12.2, 16-bit B layout)
// ---------------------------------------------------------------------------
__global__ void pack_w_bf16(const float* __restrict__ W, __bf16* __restrict__ Wp,
                            int fi, int fo) {
    const int ntN   = fo >> 4;
    const int total = (fi >> 5) * ntN * 32;
    const int t = blockIdx.x * blockDim.x + threadIdx.x;
    if (t >= total) return;
    const int lane = t & 31;
    const int tN   = (t >> 5) % ntN;
    const int kb   = (t >> 5) / ntN;
    const int half = lane >> 4, mn = lane & 15;
    const int col  = tN * 16 + mn;
    __bf16* o = Wp + (size_t)t * 16;
#pragma unroll
    for (int e = 0; e < 16; ++e) {
        const int k = kb * 32 + e + 8 * half + ((e >= 8) ? 8 : 0);
        o[e] = (__bf16)W[(size_t)k * fo + col];
    }
}

// ---------------------------------------------------------------------------
// out[r, c] = bias[c]   (initializes the SpMM accumulation target)
// ---------------------------------------------------------------------------
__global__ void bias_init(float* __restrict__ h, const float* __restrict__ b,
                          size_t total, int fo) {
    size_t i = (size_t)blockIdx.x * blockDim.x + threadIdx.x;
    if (i < total) h[i] = b[i & (size_t)(fo - 1)];   // fo is a power of two
}

// ---------------------------------------------------------------------------
// Build one 16x32 A fragment (rows row0..row0+15, K = kk..kk+31) in bf16,
// with optional packed-bf16 ReLU.
// ---------------------------------------------------------------------------
template <int RELU>
__device__ __forceinline__ v16bf load_a_frag(const float* __restrict__ arow,
                                             int base0) {
    float4 p0 = *(const float4*)(arow + base0);
    float4 p1 = *(const float4*)(arow + base0 + 4);
    float4 q0 = *(const float4*)(arow + base0 + 16);
    float4 q1 = *(const float4*)(arow + base0 + 20);
    v16bf a;
    a[0]  = (__bf16)p0.x; a[1]  = (__bf16)p0.y; a[2]  = (__bf16)p0.z; a[3]  = (__bf16)p0.w;
    a[4]  = (__bf16)p1.x; a[5]  = (__bf16)p1.y; a[6]  = (__bf16)p1.z; a[7]  = (__bf16)p1.w;
    a[8]  = (__bf16)q0.x; a[9]  = (__bf16)q0.y; a[10] = (__bf16)q0.z; a[11] = (__bf16)q0.w;
    a[12] = (__bf16)q1.x; a[13] = (__bf16)q1.y; a[14] = (__bf16)q1.z; a[15] = (__bf16)q1.w;
    if (RELU) {
        v8u au = __builtin_bit_cast(v8u, a);
#pragma unroll
        for (int j = 0; j < 8; ++j) au[j] = pk_relu_bf16(au[j]);
        a = __builtin_bit_cast(v16bf, au);
    }
    return a;
}

// ---------------------------------------------------------------------------
// S = act(H) @ W — one wave per 32(M) x (16*NT)(N) output strip.
//   MT=2 M-tiles x NT N-tiles: each B fragment feeds 2 wmma, each A feeds NT.
//   Per 32-K step: 8 A-loads + 2*NT B-loads (b128) -> 2*NT wmma.
// ---------------------------------------------------------------------------
template <int RELU, int FO>
__global__ void __launch_bounds__(32)
gemm_wmma_bf16(const float* __restrict__ H, const __bf16* __restrict__ Wp,
               float* __restrict__ S, int n, int fi) {
    constexpr int NTN = FO >> 4;                  // N-tiles total
    constexpr int NT  = (NTN >= 4) ? 4 : NTN;     // N-tiles per wave (divides NTN)
    constexpr int MT  = 2;                        // M-tiles per wave

    const int lane = threadIdx.x & 31;
    const int half = lane >> 4;
    const int mn   = lane & 15;

    const int tN0  = blockIdx.x * NT;
    const int row0 = blockIdx.y << 5;             // 32 rows per wave
    if (row0 >= n) return;                        // uniform

    const float* arow[MT];
#pragma unroll
    for (int mt = 0; mt < MT; ++mt) {
        int rowA = row0 + mt * 16 + mn;
        if (rowA >= n) rowA = n - 1;              // clamp loads; stores guarded
        arow[mt] = H + (size_t)rowA * fi;
    }

    v8f acc[MT][NT];
#pragma unroll
    for (int mt = 0; mt < MT; ++mt)
#pragma unroll
        for (int nt = 0; nt < NT; ++nt) acc[mt][nt] = v8f{};

    const int laneOff = lane * 16;

    for (int kk = 0; kk < fi; kk += 32) {
        const int base0 = kk + 8 * half;
        v16bf a[MT];
#pragma unroll
        for (int mt = 0; mt < MT; ++mt)
            a[mt] = load_a_frag<RELU>(arow[mt], base0);

        const __bf16* wp = Wp + ((size_t)(kk >> 5) * NTN + tN0) * 512 + laneOff;
        v16bf b[NT];
#pragma unroll
        for (int nt = 0; nt < NT; ++nt)
            b[nt] = *(const v16bf*)(wp + nt * 512);

#pragma unroll
        for (int nt = 0; nt < NT; ++nt)           // consecutive wmma share b[nt]
#pragma unroll
            for (int mt = 0; mt < MT; ++mt)
                acc[mt][nt] = __builtin_amdgcn_wmma_f32_16x16x32_bf16(
                    false, a[mt], false, b[nt], (short)0, acc[mt][nt],
                    false, false);
    }

    if (row0 + 32 <= n) {                         // full strip: straight stores
#pragma unroll
        for (int mt = 0; mt < MT; ++mt) {
            const int rBase = row0 + mt * 16 + 8 * half;
#pragma unroll
            for (int nt = 0; nt < NT; ++nt) {
                float* sc = S + (size_t)rBase * FO + ((tN0 + nt) << 4) + mn;
#pragma unroll
                for (int r = 0; r < 8; ++r) sc[(size_t)r * FO] = acc[mt][nt][r];
            }
        }
    } else {                                      // tail (unused when N%32==0)
#pragma unroll
        for (int mt = 0; mt < MT; ++mt) {
            const int rBase = row0 + mt * 16 + 8 * half;
#pragma unroll
            for (int nt = 0; nt < NT; ++nt) {
                const int col = ((tN0 + nt) << 4) + mn;
#pragma unroll
                for (int r = 0; r < 8; ++r) {
                    const int row = rBase + r;
                    if (row < n) S[(size_t)row * FO + col] = acc[mt][nt][r];
                }
            }
        }
    }
}

// ---------------------------------------------------------------------------
// out[rows[e], :] += vals[e] * S[cols[e], :]  — wave-per-edge, f32 HW atomics.
// ---------------------------------------------------------------------------
template <int FO>
__global__ void __launch_bounds__(256)
spmm_scatter(const int* __restrict__ rows, const int* __restrict__ cols,
             const float* __restrict__ vals, const float* __restrict__ S,
             float* __restrict__ out, int E) {
    const int lane   = threadIdx.x & 31;
    const int waveId = blockIdx.x * (blockDim.x >> 5) + (threadIdx.x >> 5);
    const int nWaves = gridDim.x * (blockDim.x >> 5);
    for (int e = waveId; e < E; e += nWaves) {
        const float v = vals[e];
        const float* __restrict__ src = S + (size_t)cols[e] * FO;
        float* __restrict__ dst = out + (size_t)rows[e] * FO;
#pragma unroll
        for (int j = 0; j < FO / 32; ++j) {
            const int c = lane + j * 32;
            __hip_atomic_fetch_add(&dst[c], v * src[c],
                                   __ATOMIC_RELAXED, __HIP_MEMORY_SCOPE_AGENT);
        }
    }
}

// Last layer: FO=16, only lanes 0-15 issue an atomic
template <>
__global__ void __launch_bounds__(256)
spmm_scatter<16>(const int* __restrict__ rows, const int* __restrict__ cols,
                 const float* __restrict__ vals, const float* __restrict__ S,
                 float* __restrict__ out, int E) {
    const int lane   = threadIdx.x & 31;
    const int waveId = blockIdx.x * (blockDim.x >> 5) + (threadIdx.x >> 5);
    const int nWaves = gridDim.x * (blockDim.x >> 5);
    for (int e = waveId; e < E; e += nWaves) {
        const float v = vals[e];
        if (lane < 16) {
            const float s = S[(size_t)cols[e] * 16 + lane];
            __hip_atomic_fetch_add(&out[(size_t)rows[e] * 16 + lane], v * s,
                                   __ATOMIC_RELAXED, __HIP_MEMORY_SCOPE_AGENT);
        }
    }
}

// ---------------------------------------------------------------------------
// In-place log_softmax over the 16 classes, one thread per row
// ---------------------------------------------------------------------------
__global__ void logsoftmax16(float* __restrict__ out, int n) {
    int r = blockIdx.x * blockDim.x + threadIdx.x;
    if (r >= n) return;
    float* p = out + (size_t)r * 16;
    float v[16];
    float m = -3.402823466e38f;
#pragma unroll
    for (int i = 0; i < 16; ++i) { v[i] = p[i]; m = fmaxf(m, v[i]); }
    float s = 0.f;
#pragma unroll
    for (int i = 0; i < 16; ++i) s += expf(v[i] - m);
    const float lse = m + logf(s);
#pragma unroll
    for (int i = 0; i < 16; ++i) p[i] = v[i] - lse;
}

// ---------------------------------------------------------------------------
// Shape dispatchers
// ---------------------------------------------------------------------------
static void launch_gemm(const float* H, const __bf16* Wp, float* S,
                        int n, int fi, int fo, int relu, hipStream_t stream) {
    const int ntN = fo >> 4;
    const int nt  = (ntN >= 4) ? 4 : ntN;
    dim3 gg((unsigned)(ntN / nt), (unsigned)((n + 31) / 32));
#define GEMM_CASE(FO)                                                          \
    case FO:                                                                   \
        if (relu) gemm_wmma_bf16<1, FO><<<gg, 32, 0, stream>>>(H, Wp, S, n, fi); \
        else      gemm_wmma_bf16<0, FO><<<gg, 32, 0, stream>>>(H, Wp, S, n, fi); \
        break;
    switch (fo) {
        GEMM_CASE(512) GEMM_CASE(256) GEMM_CASE(128) GEMM_CASE(64) GEMM_CASE(16)
        default: break;
    }
#undef GEMM_CASE
}

static void launch_spmm(const int* rows, const int* cols, const float* ev,
                        const float* S, float* dst, int E, int fo,
                        hipStream_t stream) {
#define SPMM_CASE(FO)                                                          \
    case FO:                                                                   \
        spmm_scatter<FO><<<dim3(2048), dim3(256), 0, stream>>>(rows, cols, ev, S, dst, E); \
        break;
    switch (fo) {
        SPMM_CASE(512) SPMM_CASE(256) SPMM_CASE(128) SPMM_CASE(64) SPMM_CASE(16)
        default: break;
    }
#undef SPMM_CASE
}

// ---------------------------------------------------------------------------
// Host-side orchestration
// Inputs: 0:x 1:edge_val 2:rows 3:cols, then (W_i, b_i) pairs at 4+2i / 5+2i
// ---------------------------------------------------------------------------
extern "C" void kernel_launch(void* const* d_in, const int* in_sizes, int n_in,
                              void* d_out, int out_size, void* d_ws, size_t ws_size,
                              hipStream_t stream) {
    static const int DIMS[10] = {128, 512, 512, 512, 512, 256, 128, 128, 64, 16};
    const int N = in_sizes[0] / DIMS[0];
    const int E = in_sizes[1];

    const float* x    = (const float*)d_in[0];
    const float* ev   = (const float*)d_in[1];
    const int*   rows = (const int*)d_in[2];
    const int*   cols = (const int*)d_in[3];

    const size_t maxH = (size_t)N * 512;
    float*  S  = (float*)d_ws;
    float*  Ha = S + maxH;
    float*  Hb = Ha + maxH;
    __bf16* Wb = (__bf16*)(Hb + maxH);

    // Pack all weights into WMMA B-fragment layout (bf16)
    __bf16* Wlayer[9];
    size_t woff = 0;
    for (int i = 0; i < 9; ++i) {
        const int fi = DIMS[i], fo = DIMS[i + 1];
        Wlayer[i] = Wb + woff;
        const int threads = (fi >> 5) * (fo >> 4) * 32;
        pack_w_bf16<<<dim3((threads + 255) / 256), dim3(256), 0, stream>>>(
            (const float*)d_in[4 + 2 * i], Wlayer[i], fi, fo);
        woff += (size_t)(fi * fo);
    }

    const float* cur = x;
    for (int i = 0; i < 9; ++i) {
        const int fi = DIMS[i], fo = DIMS[i + 1];

        launch_gemm(cur, Wlayer[i], S, N, fi, fo, i > 0 ? 1 : 0, stream);

        float* dst = (i == 8) ? (float*)d_out : ((i & 1) ? Hb : Ha);
        const size_t tot = (size_t)N * fo;
        bias_init<<<dim3((unsigned)((tot + 255) / 256)), dim3(256), 0, stream>>>(
            dst, (const float*)d_in[5 + 2 * i], tot, fo);

        launch_spmm(rows, cols, ev, S, dst, E, fo, stream);
        cur = dst;
    }

    logsoftmax16<<<dim3((unsigned)((N + 255) / 256)), dim3(256), 0, stream>>>((float*)d_out, N);
}
